// MMFIBlock_34119220199899
// MI455X (gfx1250) — matrix operand order repaired and checked by reading, not compiled, not run on hardware
//
#include <hip/hip_runtime.h>
#include <math.h>

#define Bq 4
#define Cc 256
#define Hh 128
#define Ww 128
#define HID 128
#define CLIPD 512
#define NCLS 9
#define GEMB 32

typedef __attribute__((ext_vector_type(16))) _Float16 v16h;
typedef __attribute__((ext_vector_type(8)))  float    v8f;
typedef __attribute__((ext_vector_type(4)))  float    f4;
typedef __attribute__((ext_vector_type(4)))  _Float16 h4;
typedef __attribute__((ext_vector_type(4)))  unsigned int v4u;
typedef __attribute__((ext_vector_type(4)))  int      v4i_t;
typedef __attribute__((ext_vector_type(8)))  int      v8i_t;

#if defined(__has_builtin)
#if __has_builtin(__builtin_amdgcn_tensor_load_to_lds) && __has_builtin(__builtin_amdgcn_s_wait_tensorcnt)
#define HAVE_TDM 1
#endif
#endif
#ifndef HAVE_TDM
#define HAVE_TDM 0
#endif

__device__ __forceinline__ float sigmoidf_(float z) { return 1.0f / (1.0f + __expf(-z)); }

// ---------------------------------------------------------------------------
// Pass A: CLIP patch similarity -> ws_sim[64].  grid 64, block 256
// ---------------------------------------------------------------------------
__global__ void clip_kernel(const float* __restrict__ clip_x,
                            const float* __restrict__ text,
                            const float* __restrict__ lin_w,
                            const float* __restrict__ lin_b,
                            const float* __restrict__ c1w,
                            const float* __restrict__ c1b,
                            const float* __restrict__ c2w,
                            const float* __restrict__ c2b,
                            float* __restrict__ ws_sim) {
  __shared__ float sfeat[256];
  __shared__ float sfeat2[512];
  __shared__ float red[256];
  __shared__ float ssim[16];
  __shared__ float stextn[16];
  __shared__ float sh1[GEMB];

  const int bp = blockIdx.x;
  const int b = bp >> 4, p = bp & 15;
  const int pr = p >> 2, pc = p & 3;
  const int tid = threadIdx.x;

  // feat[d] = mean over 256 channels and 4 adjacent pixels (b128 loads)
  {
    const int i = tid >> 3, jg = tid & 7;
    const int gh = pr * 32 + i;
    const int gwb = pc * 32 + jg * 4;
    float s = 0.f;
#pragma unroll 4
    for (int c = 0; c < Cc; ++c) {
      const f4 v = *reinterpret_cast<const f4*>(
          clip_x + (((size_t)b * Cc + c) * Hh + gh) * Ww + gwb);
      s += v[0] + v[1] + v[2] + v[3];
    }
    sfeat[tid] = s * (1.0f / 1024.0f);
  }
  __syncthreads();

  for (int o = tid; o < CLIPD; o += 256) {
    float s = lin_b[o];
    const float* wr = lin_w + (size_t)o * 256;
#pragma unroll 4
    for (int d = 0; d < 256; ++d) s += sfeat[d] * wr[d];
    sfeat2[o] = s;
  }
  __syncthreads();

  red[tid] = sfeat2[tid] * sfeat2[tid] + sfeat2[tid + 256] * sfeat2[tid + 256];
  __syncthreads();
  for (int s = 128; s > 0; s >>= 1) {
    if (tid < s) red[tid] += red[tid + s];
    __syncthreads();
  }
  const float fscale = 1.0f / fmaxf(sqrtf(red[0]), 1e-8f);

  if (tid < NCLS) {
    float s = 0.f;
    const float* tr = text + (size_t)tid * CLIPD;
#pragma unroll 4
    for (int j = 0; j < CLIPD; ++j) s += tr[j] * tr[j];
    stextn[tid] = 1.0f / fmaxf(sqrtf(s), 1e-8f);
  }
  __syncthreads();

  if (tid < NCLS) {
    float s = 0.f;
    const float* tr = text + (size_t)tid * CLIPD;
#pragma unroll 4
    for (int j = 0; j < CLIPD; ++j) s += sfeat2[j] * tr[j];
    ssim[tid] = s * fscale * stextn[tid];
  }
  __syncthreads();

  if (tid < GEMB) {
    float s = c1b[tid];
    for (int k = 0; k < NCLS; ++k) s += ssim[k] * c1w[(tid * NCLS + k) * 9 + 4];
    sh1[tid] = fmaxf(s, 0.f);
  }
  __syncthreads();

  if (tid == 0) {
    float s = c2b[0];
    for (int g = 0; g < GEMB; ++g) s += sh1[g] * c2w[g];
    ws_sim[bp] = sigmoidf_(s);
  }
}

// ---------------------------------------------------------------------------
// Pass B: CBAM channel + spatial attention.  grid 128 (br*64+bp), block 256
// ---------------------------------------------------------------------------
__global__ void cbam_kernel(const float* __restrict__ x1,
                            const float* __restrict__ x2,
                            const float* __restrict__ fc0_w1, const float* __restrict__ fc0_b1,
                            const float* __restrict__ fc0_w2, const float* __restrict__ fc0_b2,
                            const float* __restrict__ fc1_w1, const float* __restrict__ fc1_b1,
                            const float* __restrict__ fc1_w2, const float* __restrict__ fc1_b2,
                            const float* __restrict__ mlp0_w1, const float* __restrict__ mlp0_b1,
                            const float* __restrict__ mlp0_w2, const float* __restrict__ mlp0_b2,
                            const float* __restrict__ mlp1_w1, const float* __restrict__ mlp1_b1,
                            const float* __restrict__ mlp1_w2, const float* __restrict__ mlp1_b2,
                            float* __restrict__ ws_ca,   // [2][64][256]
                            float* __restrict__ ws_sp) { // [2][64][1024]
  __shared__ float sgap[256], sgmp[256];
  __shared__ float sap[1024], smp[1024];
  __shared__ float shid[HID];
  __shared__ float sw1m[288], sb1m[16], sw2m[16];

  const int br = blockIdx.x >> 6;
  const int bp = blockIdx.x & 63;
  const int b = bp >> 4, p = bp & 15;
  const int pr = p >> 2, pc = p & 3;
  const int tid = threadIdx.x;

  const float* x   = br ? x2 : x1;
  const float* w1f = br ? fc1_w1 : fc0_w1;
  const float* b1f = br ? fc1_b1 : fc0_b1;
  const float* w2f = br ? fc1_w2 : fc0_w2;
  const float* b2f = br ? fc1_b2 : fc0_b2;
  const float* mw1 = br ? mlp1_w1 : mlp0_w1;
  const float* mb1 = br ? mlp1_b1 : mlp0_b1;
  const float* mw2 = br ? mlp1_w2 : mlp0_w2;
  const float  mb2 = (br ? mlp1_b2 : mlp0_b2)[0];

  if (tid < 288) sw1m[tid] = mw1[tid];
  if (tid < 16) { sb1m[tid] = mb1[tid]; sw2m[tid] = mw2[tid]; }

  // channel stats (thread == channel), vectorized row reads
  {
    const float* base = x + (((size_t)b * Cc + tid) * Hh + pr * 32) * Ww + pc * 32;
    float s = 0.f, mx = -3.4e38f;
    for (int i = 0; i < 32; ++i) {
      const float* row = base + (size_t)i * Ww;
#pragma unroll
      for (int j4 = 0; j4 < 8; ++j4) {
        const f4 v = *reinterpret_cast<const f4*>(row + j4 * 4);
        s += v[0] + v[1] + v[2] + v[3];
        mx = fmaxf(mx, fmaxf(fmaxf(v[0], v[1]), fmaxf(v[2], v[3])));
      }
    }
    sgap[tid] = s * (1.0f / 1024.0f);
    sgmp[tid] = mx;
  }

  // spatial stats: thread -> 4 pixels
  for (int q = 0; q < 4; ++q) {
    const int pix = tid * 4 + q;
    const int i = pix >> 5, j = pix & 31;
    const size_t off = (((size_t)b * Cc) * Hh + pr * 32 + i) * Ww + pc * 32 + j;
    float s = 0.f, mx = -3.4e38f;
#pragma unroll 4
    for (int c = 0; c < Cc; ++c) {
      const float v = x[off + (size_t)c * Hh * Ww];
      s += v; mx = fmaxf(mx, v);
    }
    sap[pix] = s * (1.0f / 256.0f);
    smp[pix] = mx;
  }
  __syncthreads();

  if (tid < HID) {
    float s = b1f[tid];
    const float* wr = w1f + (size_t)tid * 512;
#pragma unroll 4
    for (int j = 0; j < 256; ++j) s += sgap[j] * wr[j] + sgmp[j] * wr[256 + j];
    shid[tid] = fmaxf(s, 0.f);
  }
  __syncthreads();

  {
    float s = b2f[tid];
    const float* wr = w2f + (size_t)tid * HID;
#pragma unroll 4
    for (int h = 0; h < HID; ++h) s += shid[h] * wr[h];
    ws_ca[((size_t)br * 64 + bp) * 256 + tid] = sigmoidf_(s);
  }

  for (int q = 0; q < 4; ++q) {
    const int pix = tid * 4 + q;
    const int i = pix >> 5, j = pix & 31;
    float acc = mb2;
    for (int oc = 0; oc < 16; ++oc) {
      float s = sb1m[oc];
      for (int dy = 0; dy < 3; ++dy) {
        const int ii = i + dy - 1;
        if (ii < 0 || ii >= 32) continue;
        for (int dx = 0; dx < 3; ++dx) {
          const int jj = j + dx - 1;
          if (jj < 0 || jj >= 32) continue;
          s += sw1m[(oc * 2 + 0) * 9 + dy * 3 + dx] * sap[ii * 32 + jj]
             + sw1m[(oc * 2 + 1) * 9 + dy * 3 + dx] * smp[ii * 32 + jj];
        }
      }
      acc += fmaxf(s, 0.f) * sw2m[oc];
    }
    ws_sp[((size_t)br * 64 + bp) * 1024 + pix] = sigmoidf_(acc);
  }
}

// ---------------------------------------------------------------------------
// Pass C: fused gate MLP (WMMA f16 16x16x32) + elementwise fuse.
// grid 4096, block 256 (8 waves).  Block = 16 consecutive pixels of one row.
// A tile staged via Tensor Data Mover (TDM) when available.
// LDS layout in one 56KB arena:
//   [0,32768)      sAf  float[512][16]  (k-major: TDM writes rows seq.)
//                  ... reused after layer1 as sG float[256][16]
//   [32768,49152)  sW   f16[256][32]    weight K-chunk
//   [49152,53248)  sH   f16[16][128]    hidden activations
//   [53248,...)    biases / ca / sp / gm
// ---------------------------------------------------------------------------
__global__ void __launch_bounds__(256)
gate_fuse_kernel(const float* __restrict__ x1,
                 const float* __restrict__ x2,
                 const float* __restrict__ gate_w1, const float* __restrict__ gate_b1,
                 const float* __restrict__ gate_w2, const float* __restrict__ gate_b2,
                 const float* __restrict__ ws_sim,
                 const float* __restrict__ ws_ca,
                 const float* __restrict__ ws_sp,
                 float* __restrict__ out1,
                 float* __restrict__ out2) {
  __shared__ __align__(16) unsigned char smem[57344];
  float*    sAf = (float*)smem;                 // 512 x 16 f32 (32KB)
  float*    sG  = (float*)smem;                 // alias: 256 x 16 f32 gate vals
  _Float16* sW  = (_Float16*)(smem + 32768);    // 256 x 32 f16 (16KB)
  _Float16* sH  = (_Float16*)(smem + 49152);    // 16 x 128 f16 (4KB)
  float*    sB1  = (float*)(smem + 53248);      // 128
  float*    sB2  = sB1 + 128;                   // 256
  float*    sCa1 = sB2 + 256;                   // 256
  float*    sCa2 = sCa1 + 256;                  // 256
  float*    sSp1 = sCa2 + 256;                  // 16
  float*    sSp2 = sSp1 + 16;                   // 16
  float*    sGm  = sSp2 + 16;                   // 1

  const int tid  = threadIdx.x;
  const int wave = tid >> 5;
  const int lane = tid & 31;
  const int lm   = lane & 15;
  const int half = lane >> 4;

  const int pix_lin = blockIdx.x * 16;
  const int b   = pix_lin >> 14;
  const int rem = pix_lin & 16383;
  const int gh  = rem >> 7;
  const int gw0 = rem & 127;
  const int bp  = b * 16 + (gh >> 5) * 4 + (gw0 >> 5);
  const int pixbase = (gh & 31) * 32 + (gw0 & 31);

  // per-block constants
  if (tid < HID) sB1[tid] = gate_b1[tid];
  sB2[tid]  = gate_b2[tid];
  sCa1[tid] = ws_ca[(size_t)bp * 256 + tid];
  sCa2[tid] = ws_ca[((size_t)64 + bp) * 256 + tid];
  if (tid < 16) {
    sSp1[tid] = ws_sp[(size_t)bp * 1024 + pixbase + tid];
    sSp2[tid] = ws_sp[((size_t)64 + bp) * 1024 + pixbase + tid];
  }
  if (tid == 0) sGm[0] = ws_sim[bp];

  // ---- stage A tile: x1|x2 rows (k) x 16 pixels (m), f32, k-major ----
#if HAVE_TDM
  if (tid < 32) {
    const unsigned int lds_base = (unsigned int)(uintptr_t)sAf;
    const unsigned long long ga1 =
        (unsigned long long)(uintptr_t)(x1 + (((size_t)b * Cc) * Hh + gh) * Ww + gw0);
    const unsigned long long ga2 =
        (unsigned long long)(uintptr_t)(x2 + (((size_t)b * Cc) * Hh + gh) * Ww + gw0);
    // D# group1: data_size=4B, tensor dims huge (no OOB), tile 16 x 256,
    // row stride = H*W elements.
    const v8i_t g1 = { 0x00020000, 0, 0x10, 0x100010, 0x100, (int)(Hh * Ww), 0, 0 };
    const v4u g0a = { 1u, lds_base, (unsigned int)ga1,
                      (unsigned int)((ga1 >> 32) & 0x1FFFFFFu) | 0x80000000u };
    const v4u g0b = { 1u, lds_base + 16384u, (unsigned int)ga2,
                      (unsigned int)((ga2 >> 32) & 0x1FFFFFFu) | 0x80000000u };
    const v4i_t z4 = { 0, 0, 0, 0 };
#if __clang_major__ >= 23
    const v8i_t z8 = { 0, 0, 0, 0, 0, 0, 0, 0 };
    __builtin_amdgcn_tensor_load_to_lds(g0a, g1, z4, z4, z8, 0);
    __builtin_amdgcn_tensor_load_to_lds(g0b, g1, z4, z4, z8, 0);
#else
    __builtin_amdgcn_tensor_load_to_lds(g0a, g1, z4, z4, 0);
    __builtin_amdgcn_tensor_load_to_lds(g0b, g1, z4, z4, 0);
#endif
    __builtin_amdgcn_s_wait_tensorcnt(0);
  }
#else
  for (int q = tid; q < 2048; q += 256) {      // 512 k-rows x 4 float4 groups
    const int k = q >> 2, mg = q & 3;
    const float* src = (k < 256)
        ? (x1 + (((size_t)b * Cc + k) * Hh + gh) * Ww + gw0 + mg * 4)
        : (x2 + (((size_t)b * Cc + (k - 256)) * Hh + gh) * Ww + gw0 + mg * 4);
    *reinterpret_cast<f4*>(&sAf[k * 16 + mg * 4]) = *reinterpret_cast<const f4*>(src);
  }
#endif
  __builtin_prefetch(gate_w2 + tid * 128, 0, 1);

  // ---------------- layer 1: [16 x 512] x [512 x 128] ----------------
  v8f acc1 = {};
  const int n0 = wave * 16;
  for (int kb = 0; kb < 16; ++kb) {
    __syncthreads();
    for (int q = tid; q < 1024; q += 256) {    // 128 n-rows x 8 quads
      const int n = q >> 3, kq = q & 7;
      const f4 wv = *reinterpret_cast<const f4*>(
          gate_w1 + (size_t)n * 512 + kb * 32 + kq * 4);
      h4 hv = { (_Float16)wv[0], (_Float16)wv[1], (_Float16)wv[2], (_Float16)wv[3] };
      *reinterpret_cast<h4*>(&sW[n * 32 + kq * 4]) = hv;
    }
    __syncthreads();
    v16h a, bf;
#pragma unroll
    for (int v = 0; v < 8; ++v) {
      const int ka = (v < 4) ? (half * 8 + 2 * v) : (16 + half * 8 + 2 * (v - 4));
      const int kk = kb * 32 + ka;
      a[2 * v]     = (_Float16)sAf[kk * 16 + lm];
      a[2 * v + 1] = (_Float16)sAf[(kk + 1) * 16 + lm];
      const int kbv = ka;
      bf[2 * v]     = sW[(n0 + lm) * 32 + kbv];
      bf[2 * v + 1] = sW[(n0 + lm) * 32 + kbv + 1];
    }
    acc1 = __builtin_amdgcn_wmma_f32_16x16x32_f16(false, a, false, bf,
                                                  (short)0, acc1, false, false);
  }
  __syncthreads();
#pragma unroll
  for (int r = 0; r < 8; ++r) {
    const int m2 = (lane < 16) ? r : (r + 8);
    const int n  = n0 + lm;
    sH[m2 * 128 + n] = (_Float16)fmaxf(acc1[r] + sB1[n], 0.f);
  }

  // ---------------- layer 2: [16 x 128] x [128 x 256] ----------------
  v8f acc2a = {}, acc2b = {};
  const int n0a = wave * 32, n0b = wave * 32 + 16;
  for (int kb = 0; kb < 4; ++kb) {
    __syncthreads();
    for (int q = tid; q < 2048; q += 256) {    // 256 n-rows x 8 quads
      const int n = q >> 3, kq = q & 7;
      const f4 wv = *reinterpret_cast<const f4*>(
          gate_w2 + (size_t)n * 128 + kb * 32 + kq * 4);
      h4 hv = { (_Float16)wv[0], (_Float16)wv[1], (_Float16)wv[2], (_Float16)wv[3] };
      *reinterpret_cast<h4*>(&sW[n * 32 + kq * 4]) = hv;
    }
    __syncthreads();
    v16h a, bfa, bfb;
#pragma unroll
    for (int v = 0; v < 8; ++v) {
      const int ka = (v < 4) ? (half * 8 + 2 * v) : (16 + half * 8 + 2 * (v - 4));
      a[2 * v]     = sH[lm * 128 + kb * 32 + ka];
      a[2 * v + 1] = sH[lm * 128 + kb * 32 + ka + 1];
      const int kbv = ka;
      bfa[2 * v]     = sW[(n0a + lm) * 32 + kbv];
      bfa[2 * v + 1] = sW[(n0a + lm) * 32 + kbv + 1];
      bfb[2 * v]     = sW[(n0b + lm) * 32 + kbv];
      bfb[2 * v + 1] = sW[(n0b + lm) * 32 + kbv + 1];
    }
    acc2a = __builtin_amdgcn_wmma_f32_16x16x32_f16(false, a, false, bfa,
                                                   (short)0, acc2a, false, false);
    acc2b = __builtin_amdgcn_wmma_f32_16x16x32_f16(false, a, false, bfb,
                                                   (short)0, acc2b, false, false);
  }

  // ---- transpose gate values through LDS (reuse A region) ----
  __syncthreads();
#pragma unroll
  for (int t = 0; t < 2; ++t) {
    const v8f accT = t ? acc2b : acc2a;
    const int n0t = wave * 32 + t * 16;
#pragma unroll
    for (int r = 0; r < 8; ++r) {
      const int m2 = (lane < 16) ? r : (r + 8);
      const int c  = n0t + lm;
      sG[c * 16 + m2] = sigmoidf_(accT[r] + sB2[c]);
    }
  }
  __syncthreads();

  // ---- fuse & write with float4 (b128) traffic ----
  const float gm = sGm[0];
  for (int e = tid; e < 1024; e += 256) {      // 256 c x 4 float4 groups
    const int c = e >> 2, mg = e & 3;
    const size_t idx = (((size_t)b * Cc + c) * Hh + gh) * Ww + gw0 + mg * 4;
    const f4 x1v = *reinterpret_cast<const f4*>(x1 + idx);
    const f4 x2v = *reinterpret_cast<const f4*>(x2 + idx);
    f4 o1, o2;
#pragma unroll
    for (int q = 0; q < 4; ++q) {
      const int m = mg * 4 + q;
      const float g = sG[c * 16 + m];
      o1[q] = x2v[q] * (sCa1[c] + sSp1[m]) * g * gm + x1v[q];
      o2[q] = x1v[q] * (sCa2[c] + sSp2[m]) * (1.0f - g) * gm + x2v[q];
    }
    *reinterpret_cast<f4*>(out1 + idx) = o1;
    *reinterpret_cast<f4*>(out2 + idx) = o2;
  }
}

// ---------------------------------------------------------------------------
extern "C" void kernel_launch(void* const* d_in, const int* in_sizes, int n_in,
                              void* d_out, int out_size, void* d_ws, size_t ws_size,
                              hipStream_t stream) {
  const float* x1     = (const float*)d_in[0];
  const float* x2     = (const float*)d_in[1];
  const float* clip_x = (const float*)d_in[2];
  const float* text   = (const float*)d_in[3];
  const float* lin_w  = (const float*)d_in[4];
  const float* lin_b  = (const float*)d_in[5];
  const float* c1w    = (const float*)d_in[6];
  const float* c1b    = (const float*)d_in[7];
  const float* c2w    = (const float*)d_in[8];
  const float* c2b    = (const float*)d_in[9];
  const float* fc0_w1 = (const float*)d_in[10];
  const float* fc0_b1 = (const float*)d_in[11];
  const float* fc0_w2 = (const float*)d_in[12];
  const float* fc0_b2 = (const float*)d_in[13];
  const float* fc1_w1 = (const float*)d_in[14];
  const float* fc1_b1 = (const float*)d_in[15];
  const float* fc1_w2 = (const float*)d_in[16];
  const float* fc1_b2 = (const float*)d_in[17];
  const float* m0w1   = (const float*)d_in[18];
  const float* m0b1   = (const float*)d_in[19];
  const float* m0w2   = (const float*)d_in[20];
  const float* m0b2   = (const float*)d_in[21];
  const float* m1w1   = (const float*)d_in[22];
  const float* m1b1   = (const float*)d_in[23];
  const float* m1w2   = (const float*)d_in[24];
  const float* m1b2   = (const float*)d_in[25];
  const float* gate_w1 = (const float*)d_in[26];
  const float* gate_b1 = (const float*)d_in[27];
  const float* gate_w2 = (const float*)d_in[28];
  const float* gate_b2 = (const float*)d_in[29];

  float* wsf    = (float*)d_ws;
  float* ws_sim = wsf;                       // 64
  float* ws_ca  = wsf + 64;                  // 2*64*256
  float* ws_sp  = wsf + 64 + 2 * 64 * 256;   // 2*64*1024

  float* out1 = (float*)d_out;
  float* out2 = out1 + (size_t)Bq * Cc * Hh * Ww;

  clip_kernel<<<64, 256, 0, stream>>>(clip_x, text, lin_w, lin_b,
                                      c1w, c1b, c2w, c2b, ws_sim);

  cbam_kernel<<<128, 256, 0, stream>>>(x1, x2,
                                       fc0_w1, fc0_b1, fc0_w2, fc0_b2,
                                       fc1_w1, fc1_b1, fc1_w2, fc1_b2,
                                       m0w1, m0b1, m0w2, m0b2,
                                       m1w1, m1b1, m1w2, m1b2,
                                       ws_ca, ws_sp);

  gate_fuse_kernel<<<4096, 256, 0, stream>>>(x1, x2,
                                             gate_w1, gate_b1, gate_w2, gate_b2,
                                             ws_sim, ws_ca, ws_sp,
                                             out1, out2);
}
